// BidirectionalMambaLayer_84902913507911
// MI455X (gfx1250) — matrix operand order repaired and checked by reading, not compiled, not run on hardware
//
#include <hip/hip_runtime.h>

#define D_MODEL 1024
#define D_INNER 2048
#define D_STATE 16
#define D_CONV  4
#define DT_RANK 64
#define BATCH   2
#define SEQ     1024
#define MTOT    (BATCH * SEQ)           // 2048 rows for all GEMMs
#define DBC_W   (DT_RANK + 2 * D_STATE) // 96

typedef __attribute__((ext_vector_type(16))) __bf16 bf16x16;
typedef __attribute__((ext_vector_type(8)))  __bf16 bf16x8;
typedef __attribute__((ext_vector_type(8)))  float  f32x8;
typedef __attribute__((ext_vector_type(4)))  int    i32x4;

// Pointer-to-AS-qualified-vector typedefs for the async copy builtin
typedef __attribute__((address_space(1))) i32x4* gas_i32x4_p;  // global
typedef __attribute__((address_space(3))) i32x4* las_i32x4_p;  // LDS

// CDNA5 async global->LDS copy path (guarded; falls back to ds_store staging)
#if defined(__gfx1250__) &&                                                   \
    __has_builtin(__builtin_amdgcn_global_load_async_to_lds_b128) &&          \
    __has_builtin(__builtin_amdgcn_s_wait_asynccnt)
#define USE_ASYNC_COPY 1
#else
#define USE_ASYNC_COPY 0
#endif

__device__ __forceinline__ unsigned short f32_to_bf16_bits(float f) {
  union { float f; unsigned int u; } c; c.f = f;
  unsigned int u = c.u;
  if ((u & 0x7fffffffu) > 0x7f800000u) return (unsigned short)((u >> 16) | 0x0040u); // NaN
  u += 0x7fffu + ((u >> 16) & 1u);   // round to nearest even
  return (unsigned short)(u >> 16);
}

__device__ __forceinline__ float sigmoidf_dev(float v) {
  return 1.0f / (1.0f + __expf(-v));
}

__device__ __forceinline__ bf16x16 cat8(bf16x8 lo, bf16x8 hi) {
  return __builtin_shufflevector(lo, hi, 0, 1, 2, 3, 4, 5, 6, 7,
                                 8, 9, 10, 11, 12, 13, 14, 15);
}

// ---------------------------------------------------------------------------
// fp32 -> bf16 bulk convert
// ---------------------------------------------------------------------------
__global__ void __launch_bounds__(256)
k_f32_to_bf16(const float* __restrict__ src, unsigned short* __restrict__ dst, int n) {
  int t = blockIdx.x * blockDim.x + threadIdx.x;
  if (t < n) dst[t] = f32_to_bf16_bits(src[t]);
}

// ---------------------------------------------------------------------------
// x (B,L,D_MODEL) fp32 -> bf16, optionally time-reversed (for backward dir)
// ---------------------------------------------------------------------------
__global__ void __launch_bounds__(256)
k_prep_x(const float* __restrict__ x, unsigned short* __restrict__ xb, int reverse) {
  int t    = blockIdx.x * blockDim.x + threadIdx.x;   // B*SEQ*D_MODEL threads
  int dcol = t & (D_MODEL - 1);
  int bl   = t / D_MODEL;
  int b    = bl / SEQ;
  int l    = bl - b * SEQ;
  int ls   = reverse ? (SEQ - 1 - l) : l;
  xb[t] = f32_to_bf16_bits(x[((size_t)b * SEQ + ls) * D_MODEL + dcol]);
}

// ---------------------------------------------------------------------------
// Simple bf16 WMMA GEMM (one 16x16 tile per wave) — used only for N=96.
// C[M x N] (fp32) = A[M x K] * B[N x K]^T
// ---------------------------------------------------------------------------
__global__ void __launch_bounds__(256)
k_gemm_bf16(const unsigned short* __restrict__ A, int lda,
            const unsigned short* __restrict__ B, int ldb,
            float* __restrict__ C, int ldc, int K) {
  const int lane = threadIdx.x & 31;
  const int wave = threadIdx.x >> 5;
  const int hl   = lane >> 4;
  const int idx  = lane & 15;
  const int tileN = blockIdx.x * 16;
  const int tileM = blockIdx.y * 128 + wave * 16;

  const unsigned short* Ar = A + (size_t)(tileM + idx) * lda;
  const unsigned short* Br = B + (size_t)(tileN + idx) * ldb;

  f32x8 acc = {0.f, 0.f, 0.f, 0.f, 0.f, 0.f, 0.f, 0.f};
  for (int k = 0; k < K; k += 32) {
    const int ka = k + hl * 8;    // A 16x32: lane<16 K{k..k+7,k+16..k+23}
    bf16x8 a_lo = *reinterpret_cast<const bf16x8*>(Ar + ka);
    bf16x8 a_hi = *reinterpret_cast<const bf16x8*>(Ar + ka + 16);
    const int kb = k + hl * 8;    // B 32x16: lane<16 K{k..k+7,k+16..k+23}
    bf16x8 b_lo = *reinterpret_cast<const bf16x8*>(Br + kb);
    bf16x8 b_hi = *reinterpret_cast<const bf16x8*>(Br + kb + 16);
    __builtin_prefetch(Ar + ka + 32, 0, 1);
    __builtin_prefetch(Br + kb + 32, 0, 1);
    acc = __builtin_amdgcn_wmma_f32_16x16x32_bf16(false, cat8(a_lo, a_hi),
                                                  false, cat8(b_lo, b_hi),
                                                  (short)0, acc, false, false);
  }
  float* Cr = C + (size_t)(tileM + hl * 8) * ldc + tileN + idx;
#pragma unroll
  for (int i = 0; i < 8; ++i) Cr[(size_t)i * ldc] = acc[i];
}

// ---------------------------------------------------------------------------
// Tiled bf16 WMMA GEMM: block = 128x128 output, LDS-staged, 8 waves,
// each wave computes a 32x64 register tile (2x4 WMMA accumulators).
// Requires: M % 128 == 0, N % 128 == 0, K % 32 == 0, lda/ldb multiples of 8.
// ---------------------------------------------------------------------------
#define BM 128
#define BN 128
#define BK 32
#define LDSW 40   // 32 + 8 bf16 pad -> 80B row stride, conflict-free b128 reads

__global__ void __launch_bounds__(256)
k_gemm_bf16_tiled(const unsigned short* __restrict__ A, int lda,
                  const unsigned short* __restrict__ B, int ldb,
                  float* __restrict__ C, int ldc, int K) {
  __shared__ unsigned short sA[BM * LDSW];
  __shared__ unsigned short sB[BN * LDSW];

  const int tid   = threadIdx.x;
  const int lane  = tid & 31;
  const int wave  = tid >> 5;
  const int hl    = lane >> 4;
  const int idx   = lane & 15;
  const int waveM = wave >> 1;          // 0..3 -> 32-row strip
  const int waveN = wave & 1;           // 0..1 -> 64-col strip
  const int blockM = blockIdx.y * BM;
  const int blockN = blockIdx.x * BN;

  const int ldRow   = tid >> 2;         // 0..63
  const int ldChunk = (tid & 3) * 8;    // 0,8,16,24 (bf16 elems)

  f32x8 acc[2][4];
#pragma unroll
  for (int mt = 0; mt < 2; ++mt)
#pragma unroll
    for (int nt = 0; nt < 4; ++nt)
      acc[mt][nt] = (f32x8){0.f, 0.f, 0.f, 0.f, 0.f, 0.f, 0.f, 0.f};

  for (int k = 0; k < K; k += BK) {
    // ---- stage A (BMxBK) and B (BNxBK) panels into LDS ----
#if USE_ASYNC_COPY
#pragma unroll
    for (int p = 0; p < 2; ++p) {
      const int row = ldRow + p * 64;
      unsigned short* ga = const_cast<unsigned short*>(
          A + (size_t)(blockM + row) * lda + k + ldChunk);
      unsigned short* gb = const_cast<unsigned short*>(
          B + (size_t)(blockN + row) * ldb + k + ldChunk);
      __builtin_amdgcn_global_load_async_to_lds_b128(
          (gas_i32x4_p)ga, (las_i32x4_p)&sA[row * LDSW + ldChunk], 0, 0);
      __builtin_amdgcn_global_load_async_to_lds_b128(
          (gas_i32x4_p)gb, (las_i32x4_p)&sB[row * LDSW + ldChunk], 0, 0);
    }
    __builtin_amdgcn_s_wait_asynccnt(0);
    __syncthreads();
#else
#pragma unroll
    for (int p = 0; p < 2; ++p) {
      const int row = ldRow + p * 64;
      bf16x8 va = *reinterpret_cast<const bf16x8*>(
          A + (size_t)(blockM + row) * lda + k + ldChunk);
      bf16x8 vb = *reinterpret_cast<const bf16x8*>(
          B + (size_t)(blockN + row) * ldb + k + ldChunk);
      *reinterpret_cast<bf16x8*>(&sA[row * LDSW + ldChunk]) = va;
      *reinterpret_cast<bf16x8*>(&sB[row * LDSW + ldChunk]) = vb;
    }
    __syncthreads();
#endif

    // ---- WMMA compute from LDS ----
    bf16x16 aop[2], bop[4];
#pragma unroll
    for (int mt = 0; mt < 2; ++mt) {
      const unsigned short* r = &sA[(waveM * 32 + mt * 16 + idx) * LDSW + hl * 8];
      aop[mt] = cat8(*reinterpret_cast<const bf16x8*>(r),
                     *reinterpret_cast<const bf16x8*>(r + 16));
    }
#pragma unroll
    for (int nt = 0; nt < 4; ++nt) {
      const unsigned short* r = &sB[(waveN * 64 + nt * 16 + idx) * LDSW + hl * 8];
      bop[nt] = cat8(*reinterpret_cast<const bf16x8*>(r),
                     *reinterpret_cast<const bf16x8*>(r + 16));
    }
#pragma unroll
    for (int mt = 0; mt < 2; ++mt)
#pragma unroll
      for (int nt = 0; nt < 4; ++nt)
        acc[mt][nt] = __builtin_amdgcn_wmma_f32_16x16x32_bf16(
            false, aop[mt], false, bop[nt], (short)0, acc[mt][nt], false, false);
    __syncthreads();
  }

  // ---- store C ----
#pragma unroll
  for (int mt = 0; mt < 2; ++mt) {
#pragma unroll
    for (int nt = 0; nt < 4; ++nt) {
      float* Cr = C + (size_t)(blockM + waveM * 32 + mt * 16 + hl * 8) * ldc +
                  blockN + waveN * 64 + nt * 16 + idx;
#pragma unroll
      for (int i = 0; i < 8; ++i) Cr[(size_t)i * ldc] = acc[mt][nt][i];
    }
  }
}

// ---------------------------------------------------------------------------
// depthwise causal conv(k=4) + bias + SiLU on xi half of xz; emits fp32 + bf16
// ---------------------------------------------------------------------------
__global__ void __launch_bounds__(256)
k_conv_silu(const float* __restrict__ xz, const float* __restrict__ cw,
            const float* __restrict__ cb, float* __restrict__ xi_f32,
            unsigned short* __restrict__ xi_bf) {
  int t  = blockIdx.x * blockDim.x + threadIdx.x;   // B*SEQ*D_INNER threads
  int d  = t & (D_INNER - 1);
  int bl = t / D_INNER;
  int l  = bl & (SEQ - 1);
  float acc = cb[d];
#pragma unroll
  for (int j = 0; j < D_CONV; ++j) {
    int ls = l - (D_CONV - 1) + j;
    if (ls >= 0)
      acc += cw[d * D_CONV + j] *
             xz[(size_t)(bl - (D_CONV - 1) + j) * (2 * D_INNER) + d];
  }
  float s = acc * sigmoidf_dev(acc);   // SiLU
  xi_f32[t] = s;
  xi_bf[t]  = f32_to_bf16_bits(s);
}

// ---------------------------------------------------------------------------
// Selective scan: one thread per (b, d) channel, serial over L.
// Shared B/C state vectors staged in LDS per block per timestep.
// ---------------------------------------------------------------------------
__global__ void __launch_bounds__(256)
k_scan(const float* __restrict__ dt_lin, const float* __restrict__ dbc,
       const float* __restrict__ xz, const float* __restrict__ xi,
       const float* __restrict__ dt_bias, const float* __restrict__ A_log,
       const float* __restrict__ Dp, unsigned short* __restrict__ y_bf) {
  __shared__ float sBC[2 * D_STATE];
  const int blkPerB = D_INNER / 256;
  const int b = blockIdx.x / blkPerB;
  const int d = (blockIdx.x % blkPerB) * 256 + threadIdx.x;

  float Aneg[D_STATE], h[D_STATE];
#pragma unroll
  for (int s = 0; s < D_STATE; ++s) {
    Aneg[s] = -__expf(A_log[d * D_STATE + s]);
    h[s] = 0.f;
  }
  const float dtb = dt_bias[d];
  const float Dd  = Dp[d];

  for (int l = 0; l < SEQ; ++l) {
    const size_t row = (size_t)b * SEQ + l;
    if (threadIdx.x < 2 * D_STATE)
      sBC[threadIdx.x] = dbc[row * DBC_W + DT_RANK + threadIdx.x];
    __syncthreads();

    float dtv = dt_lin[row * D_INNER + d] + dtb;
    float dts = (dtv > 20.f) ? dtv : __logf(1.f + __expf(dtv));  // softplus
    float xv  = xi[row * D_INNER + d];
    float bx  = dts * xv;
    float y   = 0.f;
#pragma unroll
    for (int s = 0; s < D_STATE; ++s) {
      h[s] = __expf(dts * Aneg[s]) * h[s] + bx * sBC[s];
      y += h[s] * sBC[D_STATE + s];
    }
    y += xv * Dd;
    float zv = xz[row * (2 * D_INNER) + D_INNER + d];
    y *= zv * sigmoidf_dev(zv);                                  // * silu(z)
    y_bf[row * D_INNER + d] = f32_to_bf16_bits(y);
    __syncthreads();
  }
}

// ---------------------------------------------------------------------------
// d = 0.5*(sigmoid(yf) + sigmoid(yb time-reversed)); LayerNorm; + x
// ---------------------------------------------------------------------------
__global__ void __launch_bounds__(256)
k_combine(const float* __restrict__ yf, const float* __restrict__ yb,
          const float* __restrict__ x, const float* __restrict__ g,
          const float* __restrict__ beta, float* __restrict__ out) {
  __shared__ float s_sum[256];
  __shared__ float s_sq[256];
  const int bl = blockIdx.x;         // 0 .. B*SEQ-1
  const int b  = bl / SEQ;
  const int l  = bl - b * SEQ;
  const float* rf = yf + (size_t)bl * D_MODEL;
  const float* rb = yb + ((size_t)b * SEQ + (SEQ - 1 - l)) * D_MODEL;
  const float* rx = x  + (size_t)bl * D_MODEL;

  float v[4];
  float sum = 0.f, sq = 0.f;
#pragma unroll
  for (int i = 0; i < 4; ++i) {
    int c = threadIdx.x + i * 256;
    float dv = 0.5f * (sigmoidf_dev(rf[c]) + sigmoidf_dev(rb[c]));
    v[i] = dv;
    sum += dv;
    sq  += dv * dv;
  }
  s_sum[threadIdx.x] = sum;
  s_sq[threadIdx.x]  = sq;
  __syncthreads();
  for (int st = 128; st > 0; st >>= 1) {
    if (threadIdx.x < st) {
      s_sum[threadIdx.x] += s_sum[threadIdx.x + st];
      s_sq[threadIdx.x]  += s_sq[threadIdx.x + st];
    }
    __syncthreads();
  }
  const float mu  = s_sum[0] * (1.0f / D_MODEL);
  const float var = s_sq[0] * (1.0f / D_MODEL) - mu * mu;
  const float inv = rsqrtf(var + 1e-5f);
#pragma unroll
  for (int i = 0; i < 4; ++i) {
    int c = threadIdx.x + i * 256;
    out[(size_t)bl * D_MODEL + c] = (v[i] - mu) * inv * g[c] + beta[c] + rx[c];
  }
}

// ---------------------------------------------------------------------------
// Host driver
// ---------------------------------------------------------------------------
extern "C" void kernel_launch(void* const* d_in, const int* in_sizes, int n_in,
                              void* d_out, int out_size, void* d_ws, size_t ws_size,
                              hipStream_t stream) {
  (void)in_sizes; (void)n_in; (void)out_size; (void)ws_size;

  const float* x    = (const float*)d_in[0];
  const float* ln_g = (const float*)d_in[19];
  const float* ln_b = (const float*)d_in[20];

  char*  ws  = (char*)d_ws;
  size_t off = 0;
  auto alloc = [&](size_t bytes) -> char* {
    off = (off + 255) & ~(size_t)255;
    char* p = ws + off;
    off += bytes;
    return p;
  };

  unsigned short* wbf_in  = (unsigned short*)alloc((size_t)2 * D_INNER * D_MODEL * 2);
  unsigned short* wbf_xp  = (unsigned short*)alloc((size_t)DBC_W * D_INNER * 2);
  unsigned short* wbf_dtw = (unsigned short*)alloc((size_t)D_INNER * DT_RANK * 2);
  unsigned short* wbf_out = (unsigned short*)alloc((size_t)D_MODEL * D_INNER * 2);
  unsigned short* xbf     = (unsigned short*)alloc((size_t)MTOT * D_MODEL * 2);
  float*          xz      = (float*)alloc((size_t)MTOT * 2 * D_INNER * 4);
  float*          xi_f    = (float*)alloc((size_t)MTOT * D_INNER * 4);
  unsigned short* xi_b    = (unsigned short*)alloc((size_t)MTOT * D_INNER * 2);
  float*          dbcf    = (float*)alloc((size_t)MTOT * DBC_W * 4);
  unsigned short* dbcb    = (unsigned short*)alloc((size_t)MTOT * DBC_W * 2);
  float*          dtlin   = (float*)alloc((size_t)MTOT * D_INNER * 4);
  unsigned short* ybf16   = (unsigned short*)alloc((size_t)MTOT * D_INNER * 2);
  float*          yfbuf   = (float*)alloc((size_t)MTOT * D_MODEL * 4);
  float*          ybbuf   = (float*)alloc((size_t)MTOT * D_MODEL * 4);

  for (int dir = 0; dir < 2; ++dir) {
    const int base = 1 + dir * 9;
    const float* in_proj  = (const float*)d_in[base + 0];
    const float* conv_w   = (const float*)d_in[base + 1];
    const float* conv_b   = (const float*)d_in[base + 2];
    const float* x_proj   = (const float*)d_in[base + 3];
    const float* dt_w     = (const float*)d_in[base + 4];
    const float* dt_bi    = (const float*)d_in[base + 5];
    const float* A_log    = (const float*)d_in[base + 6];
    const float* Dp       = (const float*)d_in[base + 7];
    const float* out_proj = (const float*)d_in[base + 8];
    float* ydst = dir ? ybbuf : yfbuf;

    // bf16 conversions (x possibly reversed, plus this direction's weights)
    k_prep_x<<<MTOT * D_MODEL / 256, 256, 0, stream>>>(x, xbf, dir);
    k_f32_to_bf16<<<(2 * D_INNER * D_MODEL + 255) / 256, 256, 0, stream>>>(
        in_proj, wbf_in, 2 * D_INNER * D_MODEL);
    k_f32_to_bf16<<<(DBC_W * D_INNER + 255) / 256, 256, 0, stream>>>(
        x_proj, wbf_xp, DBC_W * D_INNER);
    k_f32_to_bf16<<<(D_INNER * DT_RANK + 255) / 256, 256, 0, stream>>>(
        dt_w, wbf_dtw, D_INNER * DT_RANK);
    k_f32_to_bf16<<<(D_MODEL * D_INNER + 255) / 256, 256, 0, stream>>>(
        out_proj, wbf_out, D_MODEL * D_INNER);

    // xz = x @ in_proj^T   (2048 x 4096 x 1024) — tiled WMMA
    k_gemm_bf16_tiled<<<dim3(2 * D_INNER / BN, MTOT / BM), 256, 0, stream>>>(
        xbf, D_MODEL, wbf_in, D_MODEL, xz, 2 * D_INNER, D_MODEL);

    // depthwise causal conv + SiLU on xi half
    k_conv_silu<<<MTOT * D_INNER / 256, 256, 0, stream>>>(xz, conv_w, conv_b, xi_f, xi_b);

    // dbc = xi @ x_proj^T  (2048 x 96 x 2048) — N=96, simple WMMA
    k_gemm_bf16<<<dim3(DBC_W / 16, MTOT / 128), 256, 0, stream>>>(
        xi_b, D_INNER, wbf_xp, D_INNER, dbcf, DBC_W, D_INNER);
    k_f32_to_bf16<<<(MTOT * DBC_W + 255) / 256, 256, 0, stream>>>(dbcf, dbcb, MTOT * DBC_W);

    // dt_lin = dbc[:, :64] @ dt_w^T  (2048 x 2048 x 64), lda = 96 — tiled WMMA
    k_gemm_bf16_tiled<<<dim3(D_INNER / BN, MTOT / BM), 256, 0, stream>>>(
        dbcb, DBC_W, wbf_dtw, DT_RANK, dtlin, D_INNER, DT_RANK);

    // selective scan -> y (bf16)
    k_scan<<<BATCH * (D_INNER / 256), 256, 0, stream>>>(
        dtlin, dbcf, xz, xi_f, dt_bi, A_log, Dp, ybf16);

    // ydst = y @ out_proj^T  (2048 x 1024 x 2048) — tiled WMMA
    k_gemm_bf16_tiled<<<dim3(D_MODEL / BN, MTOT / BM), 256, 0, stream>>>(
        ybf16, D_INNER, wbf_out, D_INNER, ydst, D_MODEL, D_INNER);
  }

  // sigmoid-mix + LayerNorm + residual
  k_combine<<<MTOT, 256, 0, stream>>>(yfbuf, ybbuf, x, ln_g, ln_b, (float*)d_out);
}
